// LossFunction_2740189135094
// MI455X (gfx1250) — matrix operand order repaired and checked by reading, not compiled, not run on hardware
//
#include <hip/hip_runtime.h>
#include <math.h>

// ---------------------------------------------------------------------------
// Types for CDNA5 WMMA (wave32, gfx1250)
// ---------------------------------------------------------------------------
typedef __attribute__((ext_vector_type(16))) __bf16 v16bf;
typedef __attribute__((ext_vector_type(8)))  float  v8f;
typedef __attribute__((ext_vector_type(4)))  unsigned int u32x4;

union FragU { v16bf v; u32x4 u[2]; };

#define NOUT     512
#define NCLS     5994
#define NPAD     6016     // 94 * 64
#define BSZ      2048
#define ROWS     4096     // BSZ * 2
#define COHORT   101
#define SCALE_C  30.0f

// AAM margin constants (MARGIN = 0.2)
#define COS_M  0.98006657784124163f
#define SIN_M  0.19866933079506122f
#define TH_C  (-0.98006657784124163f)
#define MM_C   0.039733866159012244f

// ---------------------------------------------------------------------------
// helpers
// ---------------------------------------------------------------------------
__device__ __forceinline__ float hsig(float v) {
    return fminf(fmaxf((v + 3.0f) * (1.0f / 6.0f), 0.0f), 1.0f);
}

__device__ __forceinline__ unsigned fkey(float v) {
    unsigned u = __float_as_uint(v);
    return (u & 0x80000000u) ? ~u : (u | 0x80000000u);
}
__device__ __forceinline__ float funkey(unsigned k) {
    unsigned u = (k & 0x80000000u) ? (k ^ 0x80000000u) : ~k;
    return __uint_as_float(u);
}

__device__ __forceinline__ float aam_logit(float c, bool isLab) {
    if (!isLab) return c * SCALE_C;
    float cc = fminf(fmaxf(1.0f - c * c, 0.0f), 1.0f);
    float s   = sqrtf(cc);
    float phi = c * COS_M - s * SIN_M;
    float outv = (c - TH_C > 0.0f) ? phi : (c - MM_C);
    return outv * SCALE_C;
}

// blockDim.x must be 256
__device__ __forceinline__ float blockReduceSum256(float v, float* sbuf) {
    int tid = threadIdx.x;
    sbuf[tid] = v;
    __syncthreads();
    for (int s = 128; s > 0; s >>= 1) {
        if (tid < s) sbuf[tid] += sbuf[tid + s];
        __syncthreads();
    }
    float r = sbuf[0];
    __syncthreads();
    return r;
}

// ---------------------------------------------------------------------------
// Kernel 1: L2-normalize rows and split into bf16 hi/lo pairs
// grid.x = padded row count; rows >= validRows are written as zeros
// ---------------------------------------------------------------------------
__global__ __launch_bounds__(256)
void rownorm_split(const float* __restrict__ src, int validRows,
                   __bf16* __restrict__ hi, __bf16* __restrict__ lo) {
    int r   = blockIdx.x;
    int tid = threadIdx.x;
    __shared__ float sred[256];
    __shared__ float sinv;

    float s = 0.0f;
    if (r < validRows) {
        for (int c = tid; c < NOUT; c += 256) {
            float v = src[(long)r * NOUT + c];
            s += v * v;
        }
    }
    float tot = blockReduceSum256(s, sred);
    if (tid == 0) {
        float n = sqrtf(tot);
        sinv = 1.0f / fmaxf(n, 1e-12f);
    }
    __syncthreads();
    float inv = sinv;
    for (int c = tid; c < NOUT; c += 256) {
        float v = (r < validRows) ? src[(long)r * NOUT + c] * inv : 0.0f;
        __bf16 h = (__bf16)v;
        __bf16 l = (__bf16)(v - (float)h);
        hi[(long)r * NOUT + c] = h;
        lo[(long)r * NOUT + c] = l;
    }
}

// ---------------------------------------------------------------------------
// Kernel 2: split-bf16 WMMA GEMM   C[M,N] = A[M,K] * B[N,K]^T   (K = 512)
// A row i at Ah + i*aStride  (aStride in elements; lets us stride over
// interleaved utterances).  B row j at Bh + j*bStride.
// Block = 128 threads (4 waves), block tile 64x64, wave tile 32x32.
// ---------------------------------------------------------------------------
__device__ __forceinline__ v16bf load_fragA(const __bf16* p) {
    FragU f;
    f.u[0] = *reinterpret_cast<const u32x4*>(p);        // K  .. K+7
    f.u[1] = *reinterpret_cast<const u32x4*>(p + 16);   // K+16 .. K+23
    return f.v;
}
__device__ __forceinline__ v16bf load_fragB(const __bf16* p) {
    FragU f;
    f.u[0] = *reinterpret_cast<const u32x4*>(p);        // K .. K+7
    f.u[1] = *reinterpret_cast<const u32x4*>(p + 16);   // K+16 .. K+23
    return f.v;
}

#define WMMA_BF16(a, b, c) \
    __builtin_amdgcn_wmma_f32_16x16x32_bf16(false, (a), false, (b), (short)0, (c), false, false)

__global__ __launch_bounds__(128)
void wmma_gemm_split(const __bf16* __restrict__ Ah, const __bf16* __restrict__ Al, int aStride,
                     const __bf16* __restrict__ Bh, const __bf16* __restrict__ Bl, int bStride,
                     float* __restrict__ C, int ldc) {
    const int lane = threadIdx.x & 31;
    const int wave = threadIdx.x >> 5;
    const int wm = wave >> 1;
    const int wn = wave & 1;
    const int l16  = lane & 15;
    const int lhalf = lane >> 4;

    const int rowBase = blockIdx.y * 64 + wm * 32 + l16;   // logical A row (M)
    const int colBase = blockIdx.x * 64 + wn * 32 + l16;   // logical B row (N)

    const __bf16* a0h = Ah + (long)rowBase * aStride + lhalf * 8;
    const __bf16* a0l = Al + (long)rowBase * aStride + lhalf * 8;
    const __bf16* a1h = a0h + 16l * aStride;
    const __bf16* a1l = a0l + 16l * aStride;
    const __bf16* b0h = Bh + (long)colBase * bStride + lhalf * 8;
    const __bf16* b0l = Bl + (long)colBase * bStride + lhalf * 8;
    const __bf16* b1h = b0h + 16l * bStride;
    const __bf16* b1l = b0l + 16l * bStride;

    v8f acc00, acc01, acc10, acc11;
#pragma unroll
    for (int t = 0; t < 8; t++) { acc00[t] = 0.f; acc01[t] = 0.f; acc10[t] = 0.f; acc11[t] = 0.f; }

    for (int k = 0; k < NOUT; k += 32) {
        v16bf A0h = load_fragA(a0h + k);
        v16bf A1h = load_fragA(a1h + k);
        v16bf A0l = load_fragA(a0l + k);
        v16bf A1l = load_fragA(a1l + k);
        v16bf B0h = load_fragB(b0h + k);
        v16bf B1h = load_fragB(b1h + k);
        v16bf B0l = load_fragB(b0l + k);
        v16bf B1l = load_fragB(b1l + k);

        acc00 = WMMA_BF16(A0h, B0h, WMMA_BF16(A0h, B0l, WMMA_BF16(A0l, B0h, acc00)));
        acc01 = WMMA_BF16(A0h, B1h, WMMA_BF16(A0h, B1l, WMMA_BF16(A0l, B1h, acc01)));
        acc10 = WMMA_BF16(A1h, B0h, WMMA_BF16(A1h, B0l, WMMA_BF16(A1l, B0h, acc10)));
        acc11 = WMMA_BF16(A1h, B1h, WMMA_BF16(A1h, B1l, WMMA_BF16(A1l, B1h, acc11)));
    }

    const int srow = blockIdx.y * 64 + wm * 32 + lhalf * 8;  // C row for VGPR v: srow + v
    const int scol = blockIdx.x * 64 + wn * 32 + l16;
#pragma unroll
    for (int v = 0; v < 8; v++) {
        C[(long)(srow + v)      * ldc + scol     ] = acc00[v];
        C[(long)(srow + v)      * ldc + scol + 16] = acc01[v];
        C[(long)(srow + 16 + v) * ldc + scol     ] = acc10[v];
        C[(long)(srow + 16 + v) * ldc + scol + 16] = acc11[v];
    }
}

// ---------------------------------------------------------------------------
// Kernel 3: per-row AAM CE / argmax + top-101 mean/std -> hsig stats
// one block (256 threads) per cosine row (4096 rows)
// ---------------------------------------------------------------------------
__global__ __launch_bounds__(256)
void row_stats(const float* __restrict__ cosine, const int* __restrict__ label,
               const float* __restrict__ w2p, const float* __restrict__ w3p,
               const float* __restrict__ b2p, const float* __restrict__ b3p,
               float* __restrict__ rowLossS, float* __restrict__ correct,
               float* __restrict__ hm, float* __restrict__ hs) {
    int r   = blockIdx.x;
    int tid = threadIdx.x;
    __shared__ float srow[NCLS];
    __shared__ float sred[256];
    __shared__ int   sidx[256];
    __shared__ float stgt;

    const float* row = cosine + (long)r * NPAD;
    for (int j = tid; j < NCLS; j += 256) srow[j] = row[j];
    int lab = label[r >> 1];
    __syncthreads();

    // --- pass 1: max + argmax of modified logits (first-index tie-break) ---
    float mx = -3.0e38f;
    int   mi = 0x7fffffff;
    for (int j = tid; j < NCLS; j += 256) {
        float lg = aam_logit(srow[j], j == lab);
        if (lg > mx) { mx = lg; mi = j; }
    }
    sred[tid] = mx; sidx[tid] = mi;
    __syncthreads();
    for (int s = 128; s > 0; s >>= 1) {
        if (tid < s) {
            float o = sred[tid + s]; int oi = sidx[tid + s];
            if (o > sred[tid] || (o == sred[tid] && oi < sidx[tid])) { sred[tid] = o; sidx[tid] = oi; }
        }
        __syncthreads();
    }
    float rowmax = sred[0];
    int   rowarg = sidx[0];
    __syncthreads();

    // --- pass 2: sum of exp + target logit ---
    float se = 0.0f;
    for (int j = tid; j < NCLS; j += 256) {
        float lg = aam_logit(srow[j], j == lab);
        se += expf(lg - rowmax);
        if (j == lab) stgt = lg;
    }
    float sumexp = blockReduceSum256(se, sred);
    if (tid == 0) {
        rowLossS[r] = (rowmax + logf(sumexp)) - stgt;
        correct[r]  = (rowarg == lab) ? 1.0f : 0.0f;
    }

    // --- top-101: 32-step bisection on order-preserving keys ---
    unsigned klo = 0u, khi = 0xFFFFFFFFu;
    for (int it = 0; it < 32; it++) {
        unsigned mid = (unsigned)(((unsigned long long)klo + (unsigned long long)khi + 1ull) >> 1);
        float lc = 0.0f;
        for (int j = tid; j < NCLS; j += 256)
            if (fkey(srow[j]) >= mid) lc += 1.0f;
        float cnt = blockReduceSum256(lc, sred);
        if (cnt >= (float)COHORT) klo = mid; else khi = mid - 1u;
    }
    unsigned T = klo;                 // key of the 101st largest value (exact)
    float vk = funkey(T);

    float ls = 0.0f, lq = 0.0f, lc = 0.0f;
    for (int j = tid; j < NCLS; j += 256) {
        if (fkey(srow[j]) > T) {
            float v = srow[j];
            ls += v; lq += v * v; lc += 1.0f;
        }
    }
    float sum = blockReduceSum256(ls, sred);
    float sq  = blockReduceSum256(lq, sred);
    float cg  = blockReduceSum256(lc, sred);
    if (tid == 0) {
        float rem = (float)COHORT - cg;        // ties at the kth value
        sum += rem * vk;
        sq  += rem * vk * vk;
        float mean = sum / (float)COHORT;
        float var  = (sq - sum * sum / (float)COHORT) / (float)(COHORT - 1);  // ddof=1
        float sd   = sqrtf(fmaxf(var, 0.0f));
        hm[r] = hsig(mean * w2p[0] + w3p[0]);
        hs[r] = hsig(sd   * b2p[0] + b3p[0]);
    }
}

// ---------------------------------------------------------------------------
// Kernel 4: angular-prototypical CE rows for p1 (D row) and p2 (D column)
// one block (256 threads) per i in [0,2048)
// ---------------------------------------------------------------------------
__global__ __launch_bounds__(256)
void snorm_rows(const float* __restrict__ D, const float* __restrict__ hm,
                const float* __restrict__ hs, const float* __restrict__ wp,
                const float* __restrict__ bp, float* __restrict__ rowP1,
                float* __restrict__ rowP2) {
    int i   = blockIdx.x;
    int tid = threadIdx.x;
    __shared__ float srowD[BSZ];
    __shared__ float scolD[BSZ];
    __shared__ float shm[ROWS];
    __shared__ float shs[ROWS];
    __shared__ float sred[256];

    for (int j = tid; j < BSZ; j += 256) {
        srowD[j] = D[(long)i * BSZ + j];
        scolD[j] = D[(long)j * BSZ + i];
    }
    for (int j = tid; j < ROWS; j += 256) { shm[j] = hm[j]; shs[j] = hs[j]; }
    __syncthreads();

    float wv = wp[0], bv = bp[0];
    float hme_i = shm[2 * i],     hse_i = shs[2 * i];
    float hmo_i = shm[2 * i + 1], hso_i = shs[2 * i + 1];

    // pass 1: row maxima of csm1 / csm2
    float m1 = -3.0e38f, m2 = -3.0e38f;
    for (int j = tid; j < BSZ; j += 256) {
        float d1 = srowD[j];
        float c1 = 0.5f * ((d1 - shm[2 * j + 1]) / shs[2 * j + 1] + (d1 - hme_i) / hse_i) * wv + bv;
        float d2 = scolD[j];
        float c2 = 0.5f * ((d2 - shm[2 * j]) / shs[2 * j] + (d2 - hmo_i) / hso_i) * wv + bv;
        m1 = fmaxf(m1, c1);
        m2 = fmaxf(m2, c2);
    }
    sred[tid] = m1; __syncthreads();
    for (int s = 128; s > 0; s >>= 1) { if (tid < s) sred[tid] = fmaxf(sred[tid], sred[tid + s]); __syncthreads(); }
    float max1 = sred[0]; __syncthreads();
    sred[tid] = m2; __syncthreads();
    for (int s = 128; s > 0; s >>= 1) { if (tid < s) sred[tid] = fmaxf(sred[tid], sred[tid + s]); __syncthreads(); }
    float max2 = sred[0]; __syncthreads();

    // pass 2: sums of exp
    float e1 = 0.0f, e2 = 0.0f;
    for (int j = tid; j < BSZ; j += 256) {
        float d1 = srowD[j];
        float c1 = 0.5f * ((d1 - shm[2 * j + 1]) / shs[2 * j + 1] + (d1 - hme_i) / hse_i) * wv + bv;
        float d2 = scolD[j];
        float c2 = 0.5f * ((d2 - shm[2 * j]) / shs[2 * j] + (d2 - hmo_i) / hso_i) * wv + bv;
        e1 += expf(c1 - max1);
        e2 += expf(c2 - max2);
    }
    float s1 = blockReduceSum256(e1, sred);
    float s2 = blockReduceSum256(e2, sred);

    if (tid == 0) {
        float dii = srowD[i];  // == scolD[i]
        float c1ii = 0.5f * ((dii - hmo_i) / hso_i + (dii - hme_i) / hse_i) * wv + bv;
        float c2ii = 0.5f * ((dii - hme_i) / hse_i + (dii - hmo_i) / hso_i) * wv + bv;
        rowP1[i] = (max1 + logf(s1)) - c1ii;
        rowP2[i] = (max2 + logf(s2)) - c2ii;
    }
}

// ---------------------------------------------------------------------------
// Kernel 5: final reduction into d_out = { loss, prec1 }
// ---------------------------------------------------------------------------
__global__ __launch_bounds__(256)
void finalize(const float* __restrict__ rowLossS, const float* __restrict__ correct,
              const float* __restrict__ rowP1, const float* __restrict__ rowP2,
              float* __restrict__ out) {
    int tid = threadIdx.x;
    __shared__ float sred[256];
    float s = 0.f, c = 0.f, p1 = 0.f, p2 = 0.f;
    for (int j = tid; j < ROWS; j += 256) { s += rowLossS[j]; c += correct[j]; }
    for (int j = tid; j < BSZ; j += 256)  { p1 += rowP1[j];   p2 += rowP2[j]; }
    float S  = blockReduceSum256(s,  sred);
    float C  = blockReduceSum256(c,  sred);
    float P1 = blockReduceSum256(p1, sred);
    float P2 = blockReduceSum256(p2, sred);
    if (tid == 0) {
        float nlossS = S / (float)ROWS;
        float nlossP = 0.5f * (P1 / (float)BSZ + P2 / (float)BSZ);
        out[0] = nlossS + nlossP;
        out[1] = 100.0f * C / (float)ROWS;
    }
}

// ---------------------------------------------------------------------------
// launcher
// ---------------------------------------------------------------------------
extern "C" void kernel_launch(void* const* d_in, const int* in_sizes, int n_in,
                              void* d_out, int out_size, void* d_ws, size_t ws_size,
                              hipStream_t stream) {
    const float* x      = (const float*)d_in[0];   // (2048, 2, 512)
    const float* weight = (const float*)d_in[1];   // (5994, 512)
    const float* w      = (const float*)d_in[2];
    const float* b      = (const float*)d_in[3];
    const float* w2     = (const float*)d_in[4];
    const float* w3     = (const float*)d_in[5];
    const float* b2     = (const float*)d_in[6];
    const float* b3     = (const float*)d_in[7];
    const int*   label  = (const int*)d_in[8];     // (2048,)

    char*  ws  = (char*)d_ws;
    size_t off = 0;
    auto carve = [&](size_t bytes) -> void* {
        void* p = ws + off;
        off += (bytes + 255) & ~(size_t)255;
        return p;
    };

    __bf16* Xh = (__bf16*)carve((size_t)ROWS * NOUT * 2);
    __bf16* Xl = (__bf16*)carve((size_t)ROWS * NOUT * 2);
    __bf16* Wh = (__bf16*)carve((size_t)NPAD * NOUT * 2);
    __bf16* Wl = (__bf16*)carve((size_t)NPAD * NOUT * 2);
    float*  cosbuf   = (float*)carve((size_t)ROWS * NPAD * 4);
    float*  outdot   = (float*)carve((size_t)BSZ * BSZ * 4);
    float*  hm       = (float*)carve((size_t)ROWS * 4);
    float*  hs       = (float*)carve((size_t)ROWS * 4);
    float*  rowLossS = (float*)carve((size_t)ROWS * 4);
    float*  correct  = (float*)carve((size_t)ROWS * 4);
    float*  rowP1    = (float*)carve((size_t)BSZ * 4);
    float*  rowP2    = (float*)carve((size_t)BSZ * 4);

    // 1) normalize rows, split into bf16 hi/lo (W zero-padded to 6016 rows)
    rownorm_split<<<ROWS, 256, 0, stream>>>(x, ROWS, Xh, Xl);
    rownorm_split<<<NPAD, 256, 0, stream>>>(weight, NCLS, Wh, Wl);

    // 2) cosine = Xn (4096x512) @ Wn^T  -> (4096 x 6016 padded)
    wmma_gemm_split<<<dim3(NPAD / 64, ROWS / 64), 128, 0, stream>>>(
        Xh, Xl, NOUT, Wh, Wl, NOUT, cosbuf, NPAD);

    // 3) out_dot = Xn_even (2048x512) @ Xn_odd^T -> (2048 x 2048)
    //    logical row stride 1024 selects every second physical row.
    wmma_gemm_split<<<dim3(BSZ / 64, BSZ / 64), 128, 0, stream>>>(
        Xh, Xl, 2 * NOUT, Xh + NOUT, Xl + NOUT, 2 * NOUT, outdot, BSZ);

    // 4) per-row AAM CE + top-101 stats
    row_stats<<<ROWS, 256, 0, stream>>>(cosbuf, label, w2, w3, b2, b3,
                                        rowLossS, correct, hm, hs);

    // 5) angular-prototypical CE rows (p1 over rows of D, p2 over columns)
    snorm_rows<<<BSZ, 256, 0, stream>>>(outdot, hm, hs, w, b, rowP1, rowP2);

    // 6) final scalars
    finalize<<<1, 256, 0, stream>>>(rowLossS, correct, rowP1, rowP2, (float*)d_out);
}
